// DotAttenDecoderLSTM_4097398800407
// MI455X (gfx1250) — matrix operand order repaired and checked by reading, not compiled, not run on hardware
//
#include <hip/hip_runtime.h>
#include <hip/hip_bf16.h>
#include <math.h>

// Problem dims
#define B_ 128
#define S_ 128
#define H_ 1024
#define A_ 1024
#define V_ 50000
#define NEG_ (-1e11f)

typedef __attribute__((ext_vector_type(16))) _Float16 v16h;
typedef __attribute__((ext_vector_type(8)))  _Float16 v8h;
typedef __attribute__((ext_vector_type(4)))  _Float16 v4h;
typedef __attribute__((ext_vector_type(8)))  float    v8f;
typedef __attribute__((ext_vector_type(4)))  float    v4f;

__device__ __forceinline__ float sigmoid_(float x) { return 1.0f / (1.0f + __expf(-x)); }

// ---------------------------------------------------------------------------
// Fragment loaders for V_WMMA_F32_16X16X32_F16 (wave32).
// A (16x32 MxK, f16 source, row-major): lane = half*16+l holds row M=l;
//   elements 0..15 <-> K = k0 + half*16 + 0..15
// B (32x16 KxN; source W[N,K] row-major): lane holds col N=l;
//   elements 0..15 <-> K = k0 + half*16 + 0..15
// ---------------------------------------------------------------------------
__device__ __forceinline__ v16h load_frag_a16(const _Float16* __restrict__ p) {
    v8h lo = *(const v8h*)p;          // K +0..7   (16B aligned)
    v8h hi = *(const v8h*)(p + 8);    // K +8..15
    return __builtin_shufflevector(lo, hi, 0,1,2,3,4,5,6,7,8,9,10,11,12,13,14,15);
}

// f32 weight stream, non-temporal (single-use; don't pollute 192MB L2)
__device__ __forceinline__ v16h load_frag_b_nt(const float* __restrict__ p) {
    const v4f* q = (const v4f*)p;
    v4f b0 = __builtin_nontemporal_load(q);
    v4f b1 = __builtin_nontemporal_load(q + 1);
    v4f b2 = __builtin_nontemporal_load(q + 2);
    v4f b3 = __builtin_nontemporal_load(q + 3);
    v16h f;
    f[0]=(_Float16)b0[0]; f[1]=(_Float16)b0[1]; f[2]=(_Float16)b0[2]; f[3]=(_Float16)b0[3];
    f[4]=(_Float16)b1[0]; f[5]=(_Float16)b1[1]; f[6]=(_Float16)b1[2]; f[7]=(_Float16)b1[3];
    f[8]=(_Float16)b2[0]; f[9]=(_Float16)b2[1]; f[10]=(_Float16)b2[2]; f[11]=(_Float16)b2[3];
    f[12]=(_Float16)b3[0]; f[13]=(_Float16)b3[1]; f[14]=(_Float16)b3[2]; f[15]=(_Float16)b3[3];
    return f;
}

// ---------------------------------------------------------------------------
// NARROW NT-GEMM: block = one 16-wide N tile, 8 waves = 8 M-tiles (M=128).
// Optional fused second A/W pair (LSTM gates). A in f16, W in f32 (NT loads).
// Outputs f32 (Cf) and/or f16 (Ch); act=1 -> tanh.
// ---------------------------------------------------------------------------
__global__ __launch_bounds__(256) void gemm_nt_narrow(
    const _Float16* __restrict__ A0, const float* __restrict__ W0,
    const _Float16* __restrict__ A1, const float* __restrict__ W1,
    const float* __restrict__ bias0, const float* __restrict__ bias1,
    float* __restrict__ Cf, _Float16* __restrict__ Ch, int K, int N, int act)
{
    const int wave = threadIdx.x >> 5;
    const int lane = threadIdx.x & 31;
    const int half = lane >> 4;
    const int l    = lane & 15;
    const int mrow = wave << 4;
    const int ncol = blockIdx.x << 4;

    const _Float16* a0row = A0 + (size_t)(mrow + l) * K + half * 16;
    const float*    w0row = W0 + (size_t)(ncol + l) * K + half * 16;
    const _Float16* a1row = A1 ? (A1 + (size_t)(mrow + l) * K + half * 16) : nullptr;
    const float*    w1row = W1 ? (W1 + (size_t)(ncol + l) * K + half * 16) : nullptr;

    v8f acc = {};
#pragma unroll 2
    for (int k0 = 0; k0 < K; k0 += 32) {
        __builtin_prefetch(w0row + k0 + 256, 0, 0);
        v16h af = load_frag_a16(a0row + k0);
        v16h bf = load_frag_b_nt(w0row + k0);
        acc = __builtin_amdgcn_wmma_f32_16x16x32_f16(false, af, false, bf,
                                                     (short)0, acc, false, false);
        if (A1) {
            v16h af2 = load_frag_a16(a1row + k0);
            v16h bf2 = load_frag_b_nt(w1row + k0);
            acc = __builtin_amdgcn_wmma_f32_16x16x32_f16(false, af2, false, bf2,
                                                         (short)0, acc, false, false);
        }
    }

    const int n = ncol + l;
    float bsum = 0.0f;
    if (bias0) bsum += bias0[n];
    if (bias1) bsum += bias1[n];
    const int mbase = mrow + half * 8;
#pragma unroll
    for (int e = 0; e < 8; ++e) {
        float v = acc[e] + bsum;
        if (act) v = tanhf(v);
        if (Cf) Cf[(size_t)(mbase + e) * N + n] = v;
        if (Ch) Ch[(size_t)(mbase + e) * N + n] = (_Float16)v;
    }
}

// ---------------------------------------------------------------------------
// WIDE NT-GEMM: block = 128-wide N (8 waves x 16), each wave loops all 8
// M-tiles with 8 accumulators -> every streamed weight byte loaded exactly
// once grid-wide and feeds 8 WMMAs.  For the big vocab projection.
// ---------------------------------------------------------------------------
__global__ __launch_bounds__(256) void gemm_nt_wide(
    const _Float16* __restrict__ A, const float* __restrict__ W,
    const float* __restrict__ bias, float* __restrict__ Cf, int K, int N)
{
    const int wave = threadIdx.x >> 5;
    const int lane = threadIdx.x & 31;
    const int half = lane >> 4;
    const int l    = lane & 15;
    const int ncol = (blockIdx.x << 7) + (wave << 4);
    if (ncol >= N) return;

    const float*    wrow  = W + (size_t)(ncol + l) * K + half * 16;
    const _Float16* abase = A + (size_t)l * K + half * 16;

    v8f acc[8] = {};
#pragma unroll 2
    for (int k0 = 0; k0 < K; k0 += 32) {
        __builtin_prefetch(wrow + k0 + 256, 0, 0);
        v16h bf = load_frag_b_nt(wrow + k0);
#pragma unroll
        for (int t = 0; t < 8; ++t) {
            v16h af = load_frag_a16(abase + (size_t)t * 16 * K + k0);
            acc[t] = __builtin_amdgcn_wmma_f32_16x16x32_f16(false, af, false, bf,
                                                            (short)0, acc[t], false, false);
        }
    }

    const int n = ncol + l;
    const float bsum = bias ? bias[n] : 0.0f;
#pragma unroll
    for (int t = 0; t < 8; ++t) {
        const int mbase = (t << 4) + half * 8;
#pragma unroll
        for (int e = 0; e < 8; ++e)
            Cf[(size_t)(mbase + e) * N + n] = acc[t][e] + bsum;
    }
}

// ---------------------------------------------------------------------------
// Embedding gather -> f16, plus prev_h -> f16 (A operands for gates GEMM)
// ---------------------------------------------------------------------------
__global__ __launch_bounds__(256) void embed_conv_kernel(
    const int* __restrict__ input, const float* __restrict__ emb,
    const float* __restrict__ prev_h,
    _Float16* __restrict__ x16, _Float16* __restrict__ ph16)
{
    int idx = blockIdx.x * 256 + threadIdx.x;        // over B*H
    int b = idx >> 10, k = idx & (H_ - 1);
    x16[idx]  = (_Float16)emb[(size_t)input[b] * H_ + k];
    ph16[idx] = (_Float16)prev_h[idx];
}

// ---------------------------------------------------------------------------
// LSTM elementwise (gate order i,f,g,o); h,c written straight into d_out
// ---------------------------------------------------------------------------
__global__ __launch_bounds__(256) void lstm_kernel(const float* __restrict__ gates,
                                                   const float* __restrict__ prev_c,
                                                   float* __restrict__ h_out,
                                                   float* __restrict__ c_out)
{
    int idx = blockIdx.x * 256 + threadIdx.x;        // over B*H
    int b = idx >> 10, j = idx & (H_ - 1);
    const float* g = gates + (size_t)b * (4 * H_);
    float i  = sigmoid_(g[j]);
    float f  = sigmoid_(g[H_ + j]);
    float gg = tanhf(g[2 * H_ + j]);
    float o  = sigmoid_(g[3 * H_ + j]);
    float c  = f * prev_c[idx] + i * gg;
    float h  = o * tanhf(c);
    c_out[idx] = c;
    h_out[idx] = h;
}

// ---------------------------------------------------------------------------
// Fused single-pass dot attention. One block per batch row; enc[b] (512KB f32)
// is streamed from HBM exactly once (non-temporal), staged in 256KB of LDS as
// f16 (CDNA5: 320KB LDS/WGP), scores computed during staging, then softmax and
// the context pass run entirely out of LDS.  Emits cat16 = [context, h] (f16).
// Dynamic LDS: S_*H_*2 bytes tile + S_*4 bytes scores.
// ---------------------------------------------------------------------------
__global__ __launch_bounds__(256) void atten_kernel(const float* __restrict__ enc,
                                                    const float* __restrict__ h,
                                                    _Float16* __restrict__ cat16)
{
    extern __shared__ char smem_raw[];
    _Float16* elds = (_Float16*)smem_raw;                 // [S_][H_] f16 tile
    float*    sc   = (float*)(smem_raw + (size_t)S_ * H_ * sizeof(_Float16));

    const int b    = blockIdx.x;
    const int t    = threadIdx.x;
    const int wave = t >> 5;
    const int lane = t & 31;
    const float* eb = enc + (size_t)b * S_ * H_;
    const float* hb = h   + (size_t)b * H_;

    // Phase 1: stream enc rows once (NT), stage f16 to LDS, dot with h
    for (int s = wave; s < S_; s += 8) {
        const v4f* es = (const v4f*)(eb + (size_t)s * H_);
        const v4f* hv = (const v4f*)hb;
        float sum = 0.0f;
        for (int i = lane; i < H_ / 4; i += 32) {
            v4f e = __builtin_nontemporal_load(es + i);
            v4f q = hv[i];
            sum += e[0]*q[0] + e[1]*q[1] + e[2]*q[2] + e[3]*q[3];
            v4h eh = {(_Float16)e[0], (_Float16)e[1], (_Float16)e[2], (_Float16)e[3]};
            *(v4h*)(elds + (size_t)s * H_ + 4 * i) = eh;
        }
        for (int off = 16; off > 0; off >>= 1) sum += __shfl_down(sum, off, 32);
        if (lane == 0) sc[s] = sum;
    }
    __syncthreads();

    // Phase 2: masked softmax stats (redundant per thread, all from LDS)
    float mx = -INFINITY;
    for (int s = 0; s < S_; ++s) {
        float v = sc[s]; v = (v == 0.0f) ? NEG_ : v;
        mx = fmaxf(mx, v);
    }
    float den = 0.0f;
    for (int s = 0; s < S_; ++s) {
        float v = sc[s]; v = (v == 0.0f) ? NEG_ : v;
        den += __expf(v - mx);
    }
    __syncthreads();
    if (t < S_) {
        float v = sc[t]; v = (v == 0.0f) ? NEG_ : v;
        sc[t] = __expf(v - mx) / den;
    }
    __syncthreads();

    // Phase 3: context from the LDS-resident f16 tile
    float c0 = 0.f, c1 = 0.f, c2 = 0.f, c3 = 0.f;
    for (int s = 0; s < S_; ++s) {
        float a = sc[s];
        v4h e = *(const v4h*)(elds + (size_t)s * H_ + 4 * t);
        c0 += a * (float)e[0]; c1 += a * (float)e[1];
        c2 += a * (float)e[2]; c3 += a * (float)e[3];
    }
    v4h ctx = {(_Float16)c0, (_Float16)c1, (_Float16)c2, (_Float16)c3};
    _Float16* catb = cat16 + (size_t)b * (2 * H_);
    *(v4h*)(catb + 4 * t) = ctx;                          // [:,0:1024]   context
    v4f hq = ((const v4f*)hb)[t];                         // [:,1024:2048] h
    v4h hh = {(_Float16)hq[0], (_Float16)hq[1], (_Float16)hq[2], (_Float16)hq[3]};
    *(v4h*)(catb + H_ + 4 * t) = hh;
}

// ---------------------------------------------------------------------------
// In-place log-softmax over V per row: online max/sum pass (reads hit L2 since
// the GEMM wrote logits RT) + one NT-store write pass (never re-read on device)
// ---------------------------------------------------------------------------
__global__ __launch_bounds__(256) void logsoftmax_kernel(float* __restrict__ logits)
{
    __shared__ float sm[256], ss[256];
    const int b = blockIdx.x;
    const int t = threadIdx.x;
    float* row = logits + (size_t)b * V_;

    float m = -INFINITY, s = 0.0f;
    for (int i = t; i < V_; i += 256) {
        float x  = row[i];
        float mn = fmaxf(m, x);
        s = s * __expf(m - mn) + __expf(x - mn);
        m = mn;
    }
    sm[t] = m; ss[t] = s;
    __syncthreads();
    for (int off = 128; off > 0; off >>= 1) {
        if (t < off) {
            float m2 = sm[t + off], s2 = ss[t + off];
            float mn = fmaxf(sm[t], m2);
            ss[t] = ss[t] * __expf(sm[t] - mn) + s2 * __expf(m2 - mn);
            sm[t] = mn;
        }
        __syncthreads();
    }
    float lse = sm[0] + logf(ss[0]);
    for (int i = t; i < V_; i += 256) {
        float v = row[i] - lse;
        __builtin_nontemporal_store(v, row + i);
    }
}

// ---------------------------------------------------------------------------
extern "C" void kernel_launch(void* const* d_in, const int* in_sizes, int n_in,
                              void* d_out, int out_size, void* d_ws, size_t ws_size,
                              hipStream_t stream)
{
    (void)in_sizes; (void)n_in; (void)out_size; (void)ws_size;

    const int*   input  = (const int*)  d_in[0];
    const float* prev_h = (const float*)d_in[1];
    const float* prev_c = (const float*)d_in[2];
    const float* enc    = (const float*)d_in[3];
    const float* emb    = (const float*)d_in[4];
    const float* W_ih   = (const float*)d_in[5];
    const float* W_hh   = (const float*)d_in[6];
    const float* b_ih   = (const float*)d_in[7];
    const float* b_hh   = (const float*)d_in[8];
    const float* W_out  = (const float*)d_in[9];
    const float* b_out  = (const float*)d_in[10];
    const float* W_out2 = (const float*)d_in[11];
    const float* b_out2 = (const float*)d_in[12];

    // d_out = [logits 128*50000][h 128*1024][c 128*1024]
    float* logits = (float*)d_out;
    float* h_out  = logits + (size_t)B_ * V_;
    float* c_out  = h_out  + (size_t)B_ * H_;

    // workspace: f16 activations + f32 gates (~3.25 MB)
    _Float16* x16    = (_Float16*)d_ws;                  // 128*1024
    _Float16* ph16   = x16    + (size_t)B_ * H_;         // 128*1024
    _Float16* cat16  = ph16   + (size_t)B_ * H_;         // 128*2048
    _Float16* avec16 = cat16  + (size_t)B_ * 2 * H_;     // 128*1024
    float*    gates  = (float*)(avec16 + (size_t)B_ * H_); // 128*4096 f32

    // 1. embedding gather + f16 conversion of GEMM A-operands
    embed_conv_kernel<<<dim3((B_ * H_) / 256), dim3(256), 0, stream>>>(
        input, emb, prev_h, x16, ph16);
    // 2. gates = x@W_ih^T + prev_h@W_hh^T + b_ih + b_hh   (fused dual-WMMA)
    gemm_nt_narrow<<<dim3((4 * H_) / 16), dim3(256), 0, stream>>>(
        x16, W_ih, ph16, W_hh, b_ih, b_hh, gates, nullptr, H_, 4 * H_, 0);
    // 3. LSTM elementwise -> h,c (into d_out)
    lstm_kernel<<<dim3((B_ * H_) / 256), dim3(256), 0, stream>>>(gates, prev_c, h_out, c_out);
    // 4. single-pass attention (256KB LDS stage) -> cat16 = [context, h]
    atten_kernel<<<dim3(B_), dim3(256),
                   (size_t)S_ * H_ * sizeof(_Float16) + S_ * sizeof(float), stream>>>(
        enc, h_out, cat16);
    // 5. attention_vector = tanh(cat @ W_out^T + b_out), emitted directly as f16
    gemm_nt_narrow<<<dim3(A_ / 16), dim3(256), 0, stream>>>(
        cat16, W_out, nullptr, nullptr, b_out, nullptr, nullptr, avec16, 2 * H_, A_, 1);
    // 6. logits = avec @ W_out2^T + b_out2  (wide scheme: weights read once)
    gemm_nt_wide<<<dim3((V_ / 16 + 7) / 8), dim3(256), 0, stream>>>(
        avec16, W_out2, b_out2, logits, H_, V_);
    // 7. log_softmax in place
    logsoftmax_kernel<<<dim3(B_), dim3(256), 0, stream>>>(logits);
}
